// SparseLayer_90340342104440
// MI455X (gfx1250) — matrix operand order repaired and checked by reading, not compiled, not run on hardware
//
#include <hip/hip_runtime.h>
#include <hip/hip_bf16.h>

// ---------------------------------------------------------------------------
// Problem constants (from the reference)
// ---------------------------------------------------------------------------
constexpr int B_     = 8;
constexpr int N_     = 16384;
constexpr int K_     = 64;
constexpr int R_     = 2;
constexpr int C_OUT_ = 8192;
constexpr int C_IN_  = 8192;
constexpr int CELLS  = N_ / K_;     // 256 cells per batch
constexpr int NPTS   = K_ * 8;      // 512 sample points per cell (4 corners + 2 global + 2 regional)
constexpr float EPSF   = 1e-6f;
constexpr float REGIONF = 16.0f;

typedef __attribute__((ext_vector_type(16))) _Float16 v16h;
typedef __attribute__((ext_vector_type(8)))  float    v8f;

// ---------------------------------------------------------------------------
// Deterministic hash -> uniform [0,1). (Stand-in for jax threefry streams;
// compile-only environment, distributional structure preserved.)
// ---------------------------------------------------------------------------
__device__ __forceinline__ unsigned hash_u32(unsigned x) {
    x ^= x >> 16; x *= 0x7feb352dU;
    x ^= x >> 15; x *= 0x846ca68bU;
    x ^= x >> 16;
    return x;
}
__device__ __forceinline__ float hash_uniform(unsigned seed) {
    return (float)(hash_u32(seed) >> 8) * (1.0f / 16777216.0f);
}

// ---------------------------------------------------------------------------
// y[b, co] = bias[co]
// ---------------------------------------------------------------------------
__global__ void __launch_bounds__(256) init_out_kernel(const float* __restrict__ bias,
                                                       float* __restrict__ y) {
    int i = blockIdx.x * 256 + threadIdx.x;
    if (i < B_ * C_OUT_) y[i] = bias[i % C_OUT_];
}

// ---------------------------------------------------------------------------
// One block per (b, cell). 256 threads = 8 wave32s.
// ---------------------------------------------------------------------------
__global__ void __launch_bounds__(256) sparse_layer_kernel(
    const float* __restrict__ x,       // (B, C_IN)
    const float* __restrict__ means,   // (B, N, R)
    const float* __restrict__ sigmas,  // (B, N, R)
    const float* __restrict__ values,  // (B, N)
    float* __restrict__ y)             // (B, C_OUT), pre-initialized with bias
{
    __shared__ float s_mu0[K_], s_mu1[K_];     // per-Gaussian means
    __shared__ float s_iv0[K_], s_iv1[K_];     // sqrt(1/(eps+sigma))
    __shared__ float s_val[K_];                // values
    __shared__ float s_colsum[K_];             // sum over 512 points, per k
    __shared__ float s_vs[K_];                 // values[k] / colsum[k]
    __shared__ int   s_p0[NPTS], s_p1[NPTS];   // generated integer points
    __shared__ float s_w[NPTS];                // final per-point weights

    const int cell = blockIdx.x;               // 0 .. B*CELLS-1
    const int b    = cell / CELLS;
    const int c    = cell % CELLS;
    const int tid  = threadIdx.x;

    // ---- stage Gaussian params into LDS -----------------------------------
    if (tid < K_) {
        const int k = tid;
        const size_t base = ((size_t)b * N_ + (size_t)c * K_ + k) * R_;
        s_mu0[k] = means[base + 0];
        s_mu1[k] = means[base + 1];
        s_iv0[k] = sqrtf(1.0f / (EPSF + sigmas[base + 0]));
        s_iv1[k] = sqrtf(1.0f / (EPSF + sigmas[base + 1]));
        s_val[k] = values[(size_t)b * N_ + (size_t)c * K_ + k];
        s_colsum[k] = 0.0f;
    }

    // warm the x row (32 KB) through L2/WGP$ : emits global_prefetch_b8
    const float* __restrict__ xrow = x + (size_t)b * C_IN_;
    __builtin_prefetch(xrow + (size_t)tid * 32, 0, 0);

    __syncthreads();

    // ---- generate 512 integer points per cell (2 per thread) --------------
    for (int i = tid; i < NPTS; i += 256) {
        const int k = i >> 3, slot = i & 7;
        const float m0 = s_mu0[k], m1 = s_mu1[k];
        float p0, p1;
        if (slot < 4) {
            // combos order: (T,T),(T,F),(F,T),(F,F); True -> floor
            p0 = (slot & 2) ? ceilf(m0) : floorf(m0);
            p1 = (slot & 1) ? ceilf(m1) : floorf(m1);
        } else {
            const unsigned s = ((((unsigned)b * CELLS + (unsigned)c) * K_ + (unsigned)k) * 8u + (unsigned)slot) * 2u;
            if (slot < 6) {
                p0 = floorf(hash_uniform(s + 0u) * (1.0f - EPSF) * (float)C_OUT_);
                p1 = floorf(hash_uniform(s + 1u) * (1.0f - EPSF) * (float)C_IN_);
            } else {
                const float mr0 = rintf(m0), mr1 = rintf(m1);
                float lo0 = mr0 - 0.5f * REGIONF, up0 = mr0 + 0.5f * REGIONF;
                float lo1 = mr1 - 0.5f * REGIONF, up1 = mr1 + 0.5f * REGIONF;
                lo0 = (lo0 < 0.0f) ? 0.0f : lo0;
                lo0 = (up0 > (float)C_OUT_) ? ((float)C_OUT_ - REGIONF) : lo0;
                lo1 = (lo1 < 0.0f) ? 0.0f : lo1;
                lo1 = (up1 > (float)C_IN_) ? ((float)C_IN_ - REGIONF) : lo1;
                p0 = floorf(hash_uniform(s + 0u) * (1.0f - EPSF) * REGIONF + lo0);
                p1 = floorf(hash_uniform(s + 1u) * (1.0f - EPSF) * REGIONF + lo1);
            }
        }
        p0 = fminf(fmaxf(p0, 0.0f), (float)(C_OUT_ - 1));
        p1 = fminf(fmaxf(p1, 0.0f), (float)(C_IN_ - 1));
        s_p0[i] = (int)p0;
        s_p1[i] = (int)p1;
    }
    __syncthreads();

    // ---- pass 1: column sums (normalizer over the 512-point axis) ---------
    // thread t handles Gaussian k = t&63 over a 128-point chunk.
    {
        const int k = tid & 63;
        const int chunk = tid >> 6;                 // 0..3
        const float mu0 = s_mu0[k], mu1 = s_mu1[k];
        const float iv0 = s_iv0[k], iv1 = s_iv1[k];
        float acc = 0.0f;
        const int i0 = chunk * 128;
        for (int i = i0; i < i0 + 128; ++i) {
            const float d0 = ((float)s_p0[i] - mu0) * iv0;
            const float d1 = ((float)s_p1[i] - mu1) * iv1;
            acc += fmaxf(__expf(-0.5f * (d0 * d0 + d1 * d1)), EPSF);
        }
        atomicAdd(&s_colsum[k], acc);               // ds_add_f32
    }
    __syncthreads();
    if (tid < K_) s_vs[tid] = s_val[tid] / s_colsum[tid];
    __syncthreads();

    // ---- pass 2: weights[i] = sum_k props[i,k] * vs[k] via WMMA -----------
    // Per 16-point tile: D(16x16) = A(16x32 f16) * B(32x16 f16) accumulated
    // twice to cover K=64. B columns are identical (vs replicated), so every
    // D column holds the tile's weights.
    {
        const int wave = tid >> 5;                  // 0..7 (wave32)
        const int lane = tid & 31;
        const int half = lane >> 4;                 // A/B lane half per ISA layout
        const int m    = lane & 15;                 // matrix row M

        // B as packed here (16-bit, 32x16, contiguous K like A): VGPR j, pair p, lane-half h -> K = h*16 + 2j + p = h*16 + e
        v16h bmat0, bmat1;
#pragma unroll
        for (int e = 0; e < 16; ++e) {
            bmat0[e] = (_Float16)s_vs[half * 16 + e];
            bmat1[e] = (_Float16)s_vs[32 + half * 16 + e];
        }

        for (int g = wave * 4; g < wave * 4 + 4; ++g) {   // 32 tiles / 8 waves
            const int ip = g * 16 + m;
            const float p0 = (float)s_p0[ip];
            const float p1 = (float)s_p1[ip];

            // A as packed here (16-bit, 16x32, contiguous K like B): lanes 0-15 hold K 0..15,
            // lanes 16-31 hold K 16..31 (element e -> K below).
            v16h a0, a1;
#pragma unroll
            for (int e = 0; e < 16; ++e) {
                const int ka = half * 16 + e;
                {
                    const float d0 = (p0 - s_mu0[ka]) * s_iv0[ka];
                    const float d1 = (p1 - s_mu1[ka]) * s_iv1[ka];
                    a0[e] = (_Float16)fmaxf(__expf(-0.5f * (d0 * d0 + d1 * d1)), EPSF);
                }
                const int kb = ka + 32;
                {
                    const float d0 = (p0 - s_mu0[kb]) * s_iv0[kb];
                    const float d1 = (p1 - s_mu1[kb]) * s_iv1[kb];
                    a1[e] = (_Float16)fmaxf(__expf(-0.5f * (d0 * d0 + d1 * d1)), EPSF);
                }
            }

            v8f acc = {};
            acc = __builtin_amdgcn_wmma_f32_16x16x32_f16(
                false, a0, false, bmat0, (short)0, acc, false, false);
            acc = __builtin_amdgcn_wmma_f32_16x16x32_f16(
                false, a1, false, bmat1, (short)0, acc, false, false);

            // D layout: VGPR v -> row M=v (lanes 0-15) / M=v+8 (lanes 16-31);
            // all 16 columns equal, so lanes with m==0 export the tile.
            if (m == 0) {
#pragma unroll
                for (int v = 0; v < 8; ++v)
                    s_w[g * 16 + half * 8 + v] = acc[v];
            }
        }
    }
    __syncthreads();

    // ---- gather x, scale, scatter-add into y ------------------------------
    float* __restrict__ yrow = y + (size_t)b * C_OUT_;
    for (int i = tid; i < NPTS; i += 256) {
        const float contrib = s_w[i] * xrow[s_p1[i]];
        atomicAdd(&yrow[s_p0[i]], contrib);         // global_atomic_add_f32
    }
}

// ---------------------------------------------------------------------------
// Harness entry point.
// Inputs (setup_inputs order): x, means, sigmas, values, bias. Output: (B, C_OUT) f32.
// ---------------------------------------------------------------------------
extern "C" void kernel_launch(void* const* d_in, const int* in_sizes, int n_in,
                              void* d_out, int out_size, void* d_ws, size_t ws_size,
                              hipStream_t stream) {
    (void)in_sizes; (void)n_in; (void)out_size; (void)d_ws; (void)ws_size;
    const float* x      = (const float*)d_in[0];
    const float* means  = (const float*)d_in[1];
    const float* sigmas = (const float*)d_in[2];
    const float* values = (const float*)d_in[3];
    const float* bias   = (const float*)d_in[4];
    float* y = (float*)d_out;

    init_out_kernel<<<(B_ * C_OUT_ + 255) / 256, 256, 0, stream>>>(bias, y);
    sparse_layer_kernel<<<B_ * CELLS, 256, 0, stream>>>(x, means, sigmas, values, y);
}
